// EncGausSALayer_86242943304258
// MI455X (gfx1250) — matrix operand order repaired and checked by reading, not compiled, not run on hardware
//
#include <hip/hip_runtime.h>
#include <stdint.h>

#define T_DIM 1024
#define B_DIM 8
#define C_DIM 512
#define H_DIM 8
#define D_DIM 64
#define FF_DIM 2048

typedef __attribute__((ext_vector_type(16))) __bf16 v16bf;
typedef __attribute__((ext_vector_type(8)))  float  v8f;

union AFrag { v16bf v; uint32_t d[8]; };

__device__ __forceinline__ uint16_t f2bf(float f) {
  union { float f; uint32_t u; } c; c.f = f;
  uint32_t u = c.u;
  return (uint16_t)((u + 0x7FFFu + ((u >> 16) & 1u)) >> 16);
}

// A-matrix (16x32 bf16) K-offset for VGPR i, lane-half h (contiguous: k = 16h + 2i, matches B packing)
__device__ __forceinline__ int kbA(int i, int half) {
  return 2 * i + 16 * half;
}

// ---- CDNA5 async global->LDS copy (ASYNCcnt-tracked), CDNA5 ISA ----
__device__ __forceinline__ void async_b128(void* lds_dst, const void* gsrc) {
  uint32_t off = (uint32_t)(uintptr_t)lds_dst;  // low 32 bits of flat = LDS offset
  asm volatile("global_load_async_to_lds_b128 %0, %1, off"
               :: "v"(off), "v"(gsrc)
               : "memory");
}
#if __has_builtin(__builtin_amdgcn_s_wait_asynccnt)
#define WAIT_ASYNC(n) __builtin_amdgcn_s_wait_asynccnt(n)
#else
#define WAIT_ASYNC(n) asm volatile("s_wait_asynccnt %0" :: "i"(n) : "memory")
#endif

// ---------------------------------------------------------------------------
// Pack f32 weight [Kd, N] row-major -> bf16 B-fragments, contiguous per lane.
// Fragment (kt, nt): 32 lanes x 16 bf16; b[2j+o] = W[kt*32 + 2j + o + 16*half][nt*16 + n]
// ---------------------------------------------------------------------------
__global__ void pack_w(const float* __restrict__ W, uint16_t* __restrict__ Wp,
                       int Kd, int N) {
  int gid = blockIdx.x * blockDim.x + threadIdx.x;
  int total = (Kd >> 5) * (N >> 4) * 32;
  if (gid >= total) return;
  int lane = gid & 31;
  int frag = gid >> 5;
  int nF = N >> 4;
  int kt = frag / nF, nt = frag % nF;
  int n = nt * 16 + (lane & 15);
  int half = lane >> 4;
  uint16_t* dst = Wp + (size_t)gid * 16;
#pragma unroll
  for (int e = 0; e < 16; ++e) {
    int k = kt * 32 + 2 * (e >> 1) + (e & 1) + 16 * half;
    dst[e] = f2bf(W[(size_t)k * N + n]);
  }
}

// ---------------------------------------------------------------------------
// LayerNorm: one wave per row of 512. in f32 rows (t-major), out bf16.
// remap=1: out row (b*T + t) for (b,t)-major GEMM rows.
// ---------------------------------------------------------------------------
__global__ void ln_rows(const float* __restrict__ in, const float* __restrict__ g,
                        const float* __restrict__ be, uint16_t* __restrict__ out,
                        int remap) {
  int lane = threadIdx.x & 31;
  int row = blockIdx.x * (blockDim.x >> 5) + (threadIdx.x >> 5);
  if (row >= T_DIM * B_DIM) return;
  const float* p = in + (size_t)row * C_DIM;
  float v[16];
  float s = 0.f, s2 = 0.f;
#pragma unroll
  for (int i = 0; i < 16; ++i) { v[i] = p[lane + 32 * i]; s += v[i]; s2 += v[i] * v[i]; }
#pragma unroll
  for (int msk = 16; msk >= 1; msk >>= 1) {
    s  += __shfl_xor(s,  msk, 32);
    s2 += __shfl_xor(s2, msk, 32);
  }
  float mean = s * (1.f / C_DIM);
  float var  = s2 * (1.f / C_DIM) - mean * mean;
  float rstd = rsqrtf(var + 1e-5f);
  int orow = row;
  if (remap) { int t = row >> 3, b = row & 7; orow = b * T_DIM + t; }
  uint16_t* q = out + (size_t)orow * C_DIM;
#pragma unroll
  for (int i = 0; i < 16; ++i) {
    int col = lane + 32 * i;
    q[col] = f2bf((v[i] - mean) * rstd * g[col] + be[col]);
  }
}

// ---------------------------------------------------------------------------
// Generic GEMM: Out[M,N] = A_bf16[M,Kd] @ Bpacked + bias.
// Block = 8 waves = 128 rows x 64 cols; all waves share the per-k-step B tile,
// staged in LDS by double-buffered async copies (4KB per k-step).
// mode 0: outB bf16 row-major
// mode 1: outB bf16 V-transposed [ (b*H+h)*D + d ][ t ]   (rows are b*T+t, cols h*D+d)
// mode 2: outF f32, (b,t)->(t,b) remap, + resid (x)        (fc projection)
// mode 3: outF f32 row-major + resid (x1)                  (ffn2)
// ---------------------------------------------------------------------------
__global__ void __launch_bounds__(256)
gemm_bf16(const uint16_t* __restrict__ A, const uint16_t* __restrict__ Bp,
          const float* __restrict__ bias, const float* __restrict__ resid,
          float* __restrict__ outF, uint16_t* __restrict__ outB,
          int M, int N, int Kd, int mode) {
  __shared__ __align__(16) uint16_t bstage[2][2048];  // 2 x 4KB
  const int tid  = threadIdx.x;
  const int lane = tid & 31;
  const int wv   = tid >> 5;
  const int nBlocks = N >> 6;
  const int nBlk = blockIdx.x % nBlocks;
  const int bm   = blockIdx.x / nBlocks;
  const int mTile = bm * 8 + wv;
  const int n = lane & 15, half = lane >> 4, m = lane & 15;
  const int row0 = mTile << 4;
  const int nFrag16 = N >> 4;

  v8f acc[4];
#pragma unroll
  for (int nt = 0; nt < 4; ++nt) acc[nt] = 0;

  const uint16_t* aRow = A + (size_t)(row0 + m) * Kd;
  const int kSteps = Kd >> 5;

  // prime the pipeline: stage B tile for kt=0
  async_b128(&bstage[0][tid * 8],
             Bp + ((size_t)0 * nFrag16 + (nBlk << 2)) * 512 + (size_t)tid * 8);

  for (int kt = 0; kt < kSteps; ++kt) {
    const int buf = kt & 1;
    if (kt + 1 < kSteps) {
      async_b128(&bstage[buf ^ 1][tid * 8],
                 Bp + ((size_t)(kt + 1) * nFrag16 + (nBlk << 2)) * 512 +
                     (size_t)tid * 8);
      WAIT_ASYNC(1);
    } else {
      WAIT_ASYNC(0);
    }
    __syncthreads();

    AFrag a;
#pragma unroll
    for (int i = 0; i < 8; ++i)
      a.d[i] = *(const uint32_t*)(aRow + (kt << 5) + kbA(i, half));
    // preload all 4 B fragments into distinct registers so the ds_loads
    // clause together and the 4 WMMAs issue back-to-back
    v16bf bb[4];
#pragma unroll
    for (int nt = 0; nt < 4; ++nt)
      bb[nt] = *(const v16bf*)(&bstage[buf][((nt << 5) + lane) * 16]);
#pragma unroll
    for (int nt = 0; nt < 4; ++nt)
      acc[nt] = __builtin_amdgcn_wmma_f32_16x16x32_bf16(
          false, a.v, false, bb[nt], (short)0, acc[nt], false, false);
    __syncthreads();  // protect buffer from the next-next stage's overwrite
  }

#pragma unroll
  for (int nt = 0; nt < 4; ++nt) {
    int col = (nBlk << 6) + (nt << 4) + n;
    float bv = bias ? bias[col] : 0.f;
#pragma unroll
    for (int r = 0; r < 8; ++r) {
      int row = row0 + r + 8 * half;
      float val = acc[nt][r] + bv;
      if (mode == 0) {
        outB[(size_t)row * N + col] = f2bf(val);
      } else if (mode == 1) {
        int b = row >> 10, t = row & (T_DIM - 1);
        int h = col >> 6,  d = col & 63;
        outB[((size_t)(b * H_DIM + h) * D_DIM + d) * T_DIM + t] = f2bf(val);
      } else if (mode == 2) {
        int b = row >> 10, t = row & (T_DIM - 1);
        size_t idx = ((size_t)t * B_DIM + b) * C_DIM + col;
        outF[idx] = val + resid[idx];
      } else {
        size_t idx = (size_t)row * N + col;
        outF[idx] = val + resid[idx];
      }
    }
  }
}

// ---------------------------------------------------------------------------
// Flash-style attention with Gaussian distance bias.
// Block = 128 threads = 4 waves; each wave owns 16 query rows of one (b,h).
// ---------------------------------------------------------------------------
__global__ void __launch_bounds__(128)
attn_kernel(const uint16_t* __restrict__ qb, const uint16_t* __restrict__ kbuf,
            const uint16_t* __restrict__ vt, const float* __restrict__ tao,
            uint16_t* __restrict__ ob) {
  __shared__ uint16_t plds[4][16 * 32];
  const int lane = threadIdx.x & 31;
  const int wave = threadIdx.x >> 5;
  const int n = lane & 15, half = lane >> 4, m = lane & 15;
  int id = blockIdx.x;
  int qBlk = id & 15; id >>= 4;
  int h = id & 7;
  int b = id >> 3;
  const int q0 = qBlk * 64 + wave * 16;
  float tt = tao[h];
  float tinv = 1.f / (tt * tt * tt * tt);

  const uint16_t* qp = qb + ((size_t)(b * T_DIM + q0 + m)) * C_DIM + h * D_DIM;
  AFrag aq0, aq1;
#pragma unroll
  for (int i = 0; i < 8; ++i) {
    aq0.d[i] = *(const uint32_t*)(qp + kbA(i, half));
    aq1.d[i] = *(const uint32_t*)(qp + 32 + kbA(i, half));
  }
  const uint16_t* kp = kbuf + (size_t)(b * T_DIM) * C_DIM + h * D_DIM;
  const uint16_t* vp = vt + (size_t)(b * H_DIM + h) * D_DIM * T_DIM;

  float mrow[8], lrow[8];
  v8f oacc[4];
#pragma unroll
  for (int r = 0; r < 8; ++r) { mrow[r] = -3.0e38f; lrow[r] = 0.f; }
#pragma unroll
  for (int nt = 0; nt < 4; ++nt) oacc[nt] = 0;

  for (int k0 = 0; k0 < T_DIM; k0 += 32) {
    v8f s[2];
#pragma unroll
    for (int sub = 0; sub < 2; ++sub) {
      const uint16_t* kr = kp + (size_t)(k0 + sub * 16 + n) * C_DIM;
      AFrag bk0, bk1;
#pragma unroll
      for (int j = 0; j < 8; ++j) {
        bk0.d[j] = *(const uint32_t*)(kr + 2 * j + 16 * half);
        bk1.d[j] = *(const uint32_t*)(kr + 32 + 2 * j + 16 * half);
      }
      v8f sv = 0;
      sv = __builtin_amdgcn_wmma_f32_16x16x32_bf16(false, aq0.v, false, bk0.v,
                                                   (short)0, sv, false, false);
      sv = __builtin_amdgcn_wmma_f32_16x16x32_bf16(false, aq1.v, false, bk1.v,
                                                   (short)0, sv, false, false);
      s[sub] = sv;
    }
    float lm[8];
#pragma unroll
    for (int r = 0; r < 8; ++r) {
      float qi = (float)(q0 + r + 8 * half);
#pragma unroll
      for (int sub = 0; sub < 2; ++sub) {
        float kj = (float)(k0 + sub * 16 + n);
        float diff = qi - kj;
        s[sub][r] = s[sub][r] * 0.125f - 0.5f * diff * diff * tinv;
      }
      lm[r] = fmaxf(s[0][r], s[1][r]);
    }
#pragma unroll
    for (int msk = 8; msk >= 1; msk >>= 1)
#pragma unroll
      for (int r = 0; r < 8; ++r)
        lm[r] = fmaxf(lm[r], __shfl_xor(lm[r], msk, 32));
    float rs[8];
#pragma unroll
    for (int r = 0; r < 8; ++r) {
      float mnew  = fmaxf(mrow[r], lm[r]);
      float alpha = __expf(mrow[r] - mnew);
      float p0 = __expf(s[0][r] - mnew);
      float p1 = __expf(s[1][r] - mnew);
      s[0][r] = p0; s[1][r] = p1;
      rs[r] = p0 + p1;
      mrow[r] = mnew;
      lrow[r] *= alpha;
#pragma unroll
      for (int nt = 0; nt < 4; ++nt) oacc[nt][r] *= alpha;
    }
#pragma unroll
    for (int msk = 8; msk >= 1; msk >>= 1)
#pragma unroll
      for (int r = 0; r < 8; ++r)
        rs[r] += __shfl_xor(rs[r], msk, 32);
#pragma unroll
    for (int r = 0; r < 8; ++r) lrow[r] += rs[r];

    // C-layout -> A-layout transpose of P via LDS
#pragma unroll
    for (int r = 0; r < 8; ++r) {
      plds[wave][(r + 8 * half) * 32 + n]      = f2bf(s[0][r]);
      plds[wave][(r + 8 * half) * 32 + 16 + n] = f2bf(s[1][r]);
    }
    __syncthreads();
    AFrag ap;
#pragma unroll
    for (int i = 0; i < 8; ++i)
      ap.d[i] = *(const uint32_t*)(&plds[wave][m * 32 + kbA(i, half)]);

    // preload all 4 V fragments, then chain the 4 WMMAs
    AFrag bv[4];
#pragma unroll
    for (int nt = 0; nt < 4; ++nt) {
      int d = nt * 16 + n;
#pragma unroll
      for (int j = 0; j < 8; ++j)
        bv[nt].d[j] =
            *(const uint32_t*)(vp + (size_t)d * T_DIM + k0 + 2 * j + 16 * half);
    }
#pragma unroll
    for (int nt = 0; nt < 4; ++nt)
      oacc[nt] = __builtin_amdgcn_wmma_f32_16x16x32_bf16(
          false, ap.v, false, bv[nt].v, (short)0, oacc[nt], false, false);
    __syncthreads();
  }

#pragma unroll
  for (int nt = 0; nt < 4; ++nt) {
#pragma unroll
    for (int r = 0; r < 8; ++r) {
      float val = oacc[nt][r] / lrow[r];
      int row = b * T_DIM + q0 + r + 8 * half;
      int col = h * D_DIM + nt * 16 + n;
      ob[(size_t)row * C_DIM + col] = f2bf(val);
    }
  }
}

// ---------------------------------------------------------------------------
// FFN1: 9-tap shifted GEMM, rows t-major. res = relu((sum_i h2[t+s_i] @ W_i + b) / 3)
// Tap 0 quirk: uses unshifted rows (same as center tap), per reference.
// Block = 128 rows x 64 cols, B tiles async-staged in LDS (144 k-steps).
// ---------------------------------------------------------------------------
__global__ void __launch_bounds__(256)
ffn1_kernel(const uint16_t* __restrict__ h2, const uint16_t* __restrict__ Wp,
            const float* __restrict__ bias, uint16_t* __restrict__ res) {
  __shared__ __align__(16) uint16_t bstage[2][2048];
  const int tid  = threadIdx.x;
  const int lane = tid & 31;
  const int wv   = tid >> 5;
  const int nBlk = blockIdx.x & 31;   // FF/64 = 32
  const int bm   = blockIdx.x >> 5;   // 64 row-blocks
  const int mTile = bm * 8 + wv;
  const int n = lane & 15, half = lane >> 4, m = lane & 15;

  v8f acc[4];
#pragma unroll
  for (int nt = 0; nt < 4; ++nt) acc[nt] = 0;

  const int r = (mTile << 4) + m;
  const int t = r >> 3;  // B_DIM == 8
  const int nFrag16 = FF_DIM >> 4;   // 128

  async_b128(&bstage[0][tid * 8],
             Wp + ((size_t)0 * nFrag16 + (nBlk << 2)) * 512 + (size_t)tid * 8);

  for (int idx = 0; idx < 144; ++idx) {   // ktG = tap*16 + kt
    const int buf = idx & 1;
    if (idx + 1 < 144) {
      async_b128(&bstage[buf ^ 1][tid * 8],
                 Wp + ((size_t)(idx + 1) * nFrag16 + (nBlk << 2)) * 512 +
                     (size_t)tid * 8);
      WAIT_ASYNC(1);
    } else {
      WAIT_ASYNC(0);
    }
    __syncthreads();

    const int tap = idx >> 4, kt = idx & 15;
    const int s = (tap == 0) ? 0 : tap - 4;
    const int ts = t + s;
    const bool valid = (ts >= 0) && (ts < T_DIM);
    const uint16_t* arow = h2 + (size_t)(r + s * B_DIM) * C_DIM + (kt << 5);
    AFrag a;
#pragma unroll
    for (int i = 0; i < 8; ++i)
      a.d[i] = valid ? *(const uint32_t*)(arow + kbA(i, half)) : 0u;
    v16bf bb[4];
#pragma unroll
    for (int nt = 0; nt < 4; ++nt)
      bb[nt] = *(const v16bf*)(&bstage[buf][((nt << 5) + lane) * 16]);
#pragma unroll
    for (int nt = 0; nt < 4; ++nt)
      acc[nt] = __builtin_amdgcn_wmma_f32_16x16x32_bf16(
          false, a.v, false, bb[nt], (short)0, acc[nt], false, false);
    __syncthreads();
  }

  const float sc = 0.3333333333333333f;  // 9^-0.5
#pragma unroll
  for (int nt = 0; nt < 4; ++nt) {
    int col = (nBlk << 6) + (nt << 4) + n;
    float bv = bias[col];
#pragma unroll
    for (int rr = 0; rr < 8; ++rr) {
      int row = (mTile << 4) + rr + 8 * half;
      float val = (acc[nt][rr] + bv) * sc;
      val = val > 0.f ? val : 0.f;
      res[(size_t)row * FF_DIM + col] = f2bf(val);
    }
  }
}

// ---------------------------------------------------------------------------
extern "C" void kernel_launch(void* const* d_in, const int* in_sizes, int n_in,
                              void* d_out, int out_size, void* d_ws, size_t ws_size,
                              hipStream_t stream) {
  const float* x     = (const float*)d_in[0];
  const float* ln1_g = (const float*)d_in[1];
  const float* ln1_b = (const float*)d_in[2];
  const float* wq    = (const float*)d_in[3];
  const float* wq_b  = (const float*)d_in[4];
  const float* wk    = (const float*)d_in[5];
  const float* wk_b  = (const float*)d_in[6];
  const float* wv    = (const float*)d_in[7];
  const float* wv_b  = (const float*)d_in[8];
  const float* tao   = (const float*)d_in[9];
  const float* fc_w  = (const float*)d_in[10];
  const float* fc_b  = (const float*)d_in[11];
  const float* ln2_g = (const float*)d_in[12];
  const float* ln2_b = (const float*)d_in[13];
  const float* f1_w  = (const float*)d_in[14];
  const float* f1_b  = (const float*)d_in[15];
  const float* f2_w  = (const float*)d_in[16];
  const float* f2_b  = (const float*)d_in[17];

  char* ws = (char*)d_ws;
  size_t o = 0;
  uint16_t* wqp = (uint16_t*)(ws + o); o += (size_t)512 * 512 * 2;
  uint16_t* wkp = (uint16_t*)(ws + o); o += (size_t)512 * 512 * 2;
  uint16_t* wvp = (uint16_t*)(ws + o); o += (size_t)512 * 512 * 2;
  uint16_t* fcp = (uint16_t*)(ws + o); o += (size_t)512 * 512 * 2;
  uint16_t* f1p = (uint16_t*)(ws + o); o += (size_t)4608 * 2048 * 2;
  uint16_t* f2p = (uint16_t*)(ws + o); o += (size_t)2048 * 512 * 2;
  uint16_t* hx  = (uint16_t*)(ws + o); o += (size_t)8192 * 512 * 2;   // also h2
  float*    x1  = (float*)   (ws + o); o += (size_t)8192 * 512 * 4;
  uint16_t* qbf = (uint16_t*)(ws + o); o += (size_t)8192 * 512 * 2;
  uint16_t* kbf = (uint16_t*)(ws + o); o += (size_t)8192 * 512 * 2;
  uint16_t* vtb = (uint16_t*)(ws + o); o += (size_t)8192 * 512 * 2;
  uint16_t* obf = (uint16_t*)(ws + o); o += (size_t)8192 * 512 * 2;
  uint16_t* resb = qbf;  // reuse q/k/v/o span (32 MB) once attention+fc are done

  // ---- pack weights to WMMA B-fragment layout (bf16) ----
  pack_w<<<64,   256, 0, stream>>>(wq,   wqp, 512,  512);
  pack_w<<<64,   256, 0, stream>>>(wk,   wkp, 512,  512);
  pack_w<<<64,   256, 0, stream>>>(wv,   wvp, 512,  512);
  pack_w<<<64,   256, 0, stream>>>(fc_w, fcp, 512,  512);
  pack_w<<<2304, 256, 0, stream>>>(f1_w, f1p, 4608, 2048);  // [9*512, 2048]
  pack_w<<<256,  256, 0, stream>>>(f2_w, f2p, 2048, 512);

  // ---- LN1: x[T,B,C] -> hx bf16 rows (b*T+t) ----
  ln_rows<<<1024, 256, 0, stream>>>(x, ln1_g, ln1_b, hx, 1);

  // ---- QKV projections ----
  gemm_bf16<<<512, 256, 0, stream>>>(hx, wqp, wq_b, nullptr, nullptr, qbf, 8192, 512, 512, 0);
  gemm_bf16<<<512, 256, 0, stream>>>(hx, wkp, wk_b, nullptr, nullptr, kbf, 8192, 512, 512, 0);
  gemm_bf16<<<512, 256, 0, stream>>>(hx, wvp, wv_b, nullptr, nullptr, vtb, 8192, 512, 512, 1);

  // ---- attention (flash softmax + Gaussian bias) ----
  attn_kernel<<<1024, 128, 0, stream>>>(qbf, kbf, vtb, tao, obf);

  // ---- output projection + residual, remap to [T,B,C] -> x1 ----
  gemm_bf16<<<512, 256, 0, stream>>>(obf, fcp, fc_b, x, x1, nullptr, 8192, 512, 512, 2);

  // ---- LN2: x1 -> h2 (t-major, reuse hx) ----
  ln_rows<<<1024, 256, 0, stream>>>(x1, ln2_g, ln2_b, hx, 0);

  // ---- FFN1: 9-tap shifted GEMM + relu -> res bf16 ----
  ffn1_kernel<<<2048, 256, 0, stream>>>(hx, f1p, f1_b, resb);

  // ---- FFN2 + bias + residual -> d_out ----
  gemm_bf16<<<512, 256, 0, stream>>>(resb, f2p, f2_b, x1, (float*)d_out, nullptr,
                                     8192, 512, 2048, 3);
}
